// TradeAnomalyGAT_82927228551438
// MI455X (gfx1250) — matrix operand order repaired and checked by reading, not compiled, not run on hardware
//
#include <hip/hip_runtime.h>
#include <hip/hip_bf16.h>

typedef __attribute__((ext_vector_type(16))) __bf16 v16bf;
typedef __attribute__((ext_vector_type(8)))  __bf16 v8bf;
typedef __attribute__((ext_vector_type(8)))  float  v8f;

#define NEG_SLOPE 0.2f

// ---------------------------------------------------------------- utilities
__global__ void fill_u32_kernel(unsigned* __restrict__ p, unsigned v, long long n) {
  long long i = (long long)blockIdx.x * blockDim.x + threadIdx.x;
  if (i < n) p[i] = v;
}

__global__ void f32_to_bf16_kernel(const float* __restrict__ x, __bf16* __restrict__ y,
                                   long long n) {
  long long i = (long long)blockIdx.x * blockDim.x + threadIdx.x;
  if (i < n) y[i] = (__bf16)x[i];
}

__device__ __forceinline__ unsigned enc_f(float f) {
  unsigned b = __float_as_uint(f);
  return (b & 0x80000000u) ? ~b : (b | 0x80000000u);
}
__device__ __forceinline__ float dec_f(unsigned u) {
  return __uint_as_float((u & 0x80000000u) ? (u ^ 0x80000000u) : ~u);
}

// ---------------------------------------------------------------- B pre-pack
// Pack B[K,N] (f32, row-major) into WMMA B-fragment order, bf16:
//   Bp[(((kt*NT + nt)*32 + lane)*16 + e]  where, in this code's contiguous k order (not the ISA table order), B 32x16:
//   lanes 0-15 (col=l) hold K=kb+0..15, lanes 16-31 hold K=kb+16..31.
__global__ void pack_B_kernel(const float* __restrict__ B, __bf16* __restrict__ Bp,
                              int K, int N) {
  const int t  = blockIdx.x * blockDim.x + threadIdx.x;
  const int KT = K >> 5, NT = N >> 4;
  if (t >= KT * NT * 32) return;
  const int lane = t & 31;
  const int frag = t >> 5;            // kt*NT + nt
  const int nt   = frag % NT;
  const int kt   = frag / NT;
  const int half = lane >> 4, l = lane & 15;
  const float* sp = B + (size_t)(kt * 32 + half * 16) * N + nt * 16 + l;
  __bf16* dp = Bp + (size_t)t * 16;
#pragma unroll
  for (int e = 0; e < 16; ++e) dp[e] = (__bf16)sp[(size_t)e * N];
}

// ---------------------------------------------------------------- WMMA GEMM
// C[M,N] = A[M,K] @ B[K,N]; A bf16 row-major, Bp pre-packed bf16 fragments.
// One wave computes a 16-row strip across all NT = N/16 column tiles.
// A 16x32 bf16 fragment: lanes 0-15 row=l, elems0-15 -> K+0..15;
// lanes 16-31 row=l, elems0-15 -> K+16..31 (matches packed B).
template <int NT>
__global__ __launch_bounds__(256)
void gemm_bf16_wmma(const __bf16* __restrict__ A, const __bf16* __restrict__ Bp,
                    float* __restrict__ C, int M, int N, int K) {
  const int lane = threadIdx.x & 31;
  const int wid  = blockIdx.x * (blockDim.x >> 5) + (threadIdx.x >> 5);
  const int m0   = wid * 16;
  if (m0 + 16 > M) return;
  const int half = lane >> 4;
  const int l    = lane & 15;

  const __bf16* arow = A + (size_t)(m0 + l) * K;

  v8f acc[NT];
#pragma unroll
  for (int t = 0; t < NT; ++t) acc[t] = v8f{0.f, 0.f, 0.f, 0.f, 0.f, 0.f, 0.f, 0.f};

  const int KT = K >> 5;
  for (int kt = 0; kt < KT; ++kt) {
    const int kb = kt << 5;
    const v16bf a = *(const v16bf*)(arow + kb + half * 16);       // 32B
#pragma unroll
    for (int t = 0; t < NT; ++t) {
      const v16bf b =
          *(const v16bf*)(Bp + ((size_t)(kt * NT + t) * 32 + lane) * 16);  // 32B
      acc[t] = __builtin_amdgcn_wmma_f32_16x16x32_bf16(
          false, a, false, b, (short)0, acc[t], false, false);
    }
  }

#pragma unroll
  for (int t = 0; t < NT; ++t) {
    float* cp = C + (size_t)(m0 + half * 8) * N + t * 16 + l;
#pragma unroll
    for (int j = 0; j < 8; ++j) cp[(size_t)j * N] = acc[t][j];
  }
}

// ---------------------------------------------------------------- attention dots
// as[n,h] = sum_c h[n,h,c]*a_src[h,c]; ad likewise. One wave per (node, head).
__global__ void alpha_reduce_kernel(const float* __restrict__ h,
                                    const float* __restrict__ a_src,
                                    const float* __restrict__ a_dst,
                                    float* __restrict__ as, float* __restrict__ ad,
                                    int Nn, int H) {
  const int w    = blockIdx.x * (blockDim.x >> 5) + (threadIdx.x >> 5);
  const int lane = threadIdx.x & 31;
  if (w >= Nn * H) return;
  const int n = w / H, hd = w % H;
  const float* row = h + (size_t)n * H * 64 + hd * 64;
  const float* sv  = a_src + hd * 64;
  const float* dv  = a_dst + hd * 64;
  const float v0 = row[lane], v1 = row[lane + 32];
  float s = v0 * sv[lane] + v1 * sv[lane + 32];
  float d = v0 * dv[lane] + v1 * dv[lane + 32];
#pragma unroll
  for (int m = 16; m; m >>= 1) {
    s += __shfl_xor(s, m, 32);
    d += __shfl_xor(d, m, 32);
  }
  if (lane == 0) { as[w] = s; ad[w] = d; }
}

// ---------------------------------------------------------------- edge passes
template <int H>
__global__ void edge_pass1_kernel(const int* __restrict__ src, const int* __restrict__ dst,
                                  const float* __restrict__ as, const float* __restrict__ ad,
                                  float* __restrict__ ev, unsigned* __restrict__ menc,
                                  int Eraw, int Etot) {
  const int e = blockIdx.x * blockDim.x + threadIdx.x;
  if (e >= Etot) return;
  int s, d;
  if (e < Eraw) { s = src[e]; d = dst[e]; } else { s = d = e - Eraw; }
  float sa[H], da[H];
#pragma unroll
  for (int h = 0; h < H; ++h) sa[h] = as[s * H + h];
#pragma unroll
  for (int h = 0; h < H; ++h) da[h] = ad[d * H + h];
#pragma unroll
  for (int h = 0; h < H; ++h) {
    float v = sa[h] + da[h];
    v = v > 0.f ? v : NEG_SLOPE * v;
    ev[(size_t)e * H + h] = v;
    atomicMax(&menc[d * H + h], enc_f(v));
  }
}

template <int H>
__global__ void edge_pass2_kernel(const int* __restrict__ dst, float* __restrict__ ev,
                                  const unsigned* __restrict__ menc,
                                  float* __restrict__ denom, int Eraw, int Etot) {
  const int e = blockIdx.x * blockDim.x + threadIdx.x;
  if (e >= Etot) return;
  const int d = (e < Eraw) ? dst[e] : (e - Eraw);
  float ex[H];
#pragma unroll
  for (int h = 0; h < H; ++h) {
    const float m = dec_f(menc[d * H + h]);
    ex[h] = __expf(ev[(size_t)e * H + h] - m);
  }
#pragma unroll
  for (int h = 0; h < H; ++h) {
    ev[(size_t)e * H + h] = ex[h];
    atomicAdd(&denom[d * H + h], ex[h]);
  }
}

// pass3: out[dst, h*64+c] += h[src, h*64+c] * ex/(denom[dst,h]+eps). Wave per edge,
// lanes stride the H*64 channels (coalesced gather + f32 scatter-atomics).
template <int H>
__global__ void edge_pass3_kernel(const int* __restrict__ src, const int* __restrict__ dst,
                                  const float* __restrict__ ev, const float* __restrict__ denom,
                                  const float* __restrict__ hfeat, float* __restrict__ out,
                                  int Eraw, int Etot) {
  const int w    = blockIdx.x * (blockDim.x >> 5) + (threadIdx.x >> 5);
  const int lane = threadIdx.x & 31;
  if (w >= Etot) return;
  int s, d;
  if (w < Eraw) { s = src[w]; d = dst[w]; } else { s = d = w - Eraw; }
  constexpr int C = H * 64;
#pragma unroll
  for (int cc = 0; cc < C; cc += 32) {
    const int c = cc + lane;
    const int h = c >> 6;
    const float alpha = ev[(size_t)w * H + h] / (denom[d * H + h] + 1e-16f);
    atomicAdd(&out[(size_t)d * C + c], hfeat[(size_t)s * C + c] * alpha);
  }
}

// ---------------------------------------------------------------- bias + relu
__global__ void bias_relu_kernel(float* __restrict__ x, const float* __restrict__ b,
                                 long long n, int C) {
  const long long i = (long long)blockIdx.x * blockDim.x + threadIdx.x;
  if (i >= n) return;
  const float v = x[i] + b[(int)(i % C)];
  x[i] = v > 0.f ? v : 0.f;
}

// bias + relu, writing bf16 (feeds the next WMMA GEMM directly)
__global__ void bias_relu_bf16_kernel(const float* __restrict__ x, __bf16* __restrict__ y,
                                      const float* __restrict__ b, long long n, int C) {
  const long long i = (long long)blockIdx.x * blockDim.x + threadIdx.x;
  if (i >= n) return;
  const float v = x[i] + b[(int)(i % C)];
  y[i] = (__bf16)(v > 0.f ? v : 0.f);
}

// ---------------------------------------------------------------- classifier head
// out[n,:2] = relu(h3[n,:]@Wc1 + bc1) @ Wc2 + bc2   (64 -> 32 -> 2, one thread/node)
__global__ void classifier_kernel(const float* __restrict__ h3, const float* __restrict__ Wc1,
                                  const float* __restrict__ bc1, const float* __restrict__ Wc2,
                                  const float* __restrict__ bc2, float* __restrict__ out, int Nn) {
  const int n = blockIdx.x * blockDim.x + threadIdx.x;
  if (n >= Nn) return;
  const float* r = h3 + (size_t)n * 64;
  float o0 = bc2[0], o1 = bc2[1];
#pragma unroll 4
  for (int j = 0; j < 32; ++j) {
    float acc = bc1[j];
    for (int c = 0; c < 64; ++c) acc += r[c] * Wc1[c * 32 + j];
    acc = acc > 0.f ? acc : 0.f;
    o0 += acc * Wc2[j * 2 + 0];
    o1 += acc * Wc2[j * 2 + 1];
  }
  out[(size_t)n * 2 + 0] = o0;
  out[(size_t)n * 2 + 1] = o1;
}

// ---------------------------------------------------------------- launch
static inline int cdiv_ll(long long a, long long b) { return (int)((a + b - 1) / b); }

extern "C" void kernel_launch(void* const* d_in, const int* in_sizes, int n_in,
                              void* d_out, int out_size, void* d_ws, size_t ws_size,
                              hipStream_t stream) {
  const float* x      = (const float*)d_in[0];
  const int*   ei     = (const int*)d_in[1];
  const float* W1     = (const float*)d_in[2];
  const float* a_src1 = (const float*)d_in[3];
  const float* a_dst1 = (const float*)d_in[4];
  const float* b1     = (const float*)d_in[5];
  const float* W2     = (const float*)d_in[6];
  const float* a_src2 = (const float*)d_in[7];
  const float* a_dst2 = (const float*)d_in[8];
  const float* b2     = (const float*)d_in[9];
  const float* Wc1    = (const float*)d_in[10];
  const float* bc1    = (const float*)d_in[11];
  const float* Wc2    = (const float*)d_in[12];
  const float* bc2    = (const float*)d_in[13];
  float* out = (float*)d_out;

  const int Nn   = in_sizes[0] / 128;   // 50000
  const int Eraw = in_sizes[1] / 2;     // 600000
  const int Etot = Eraw + Nn;           // with self loops
  const int H1 = 4, C1 = 256, C2 = 64, K1 = 128;

  const int* src = ei;
  const int* dst = ei + Eraw;

  // workspace layout (float units; all chunk sizes are multiples of 8 floats
  // so every bf16 sub-buffer stays 32-byte aligned)
  float* ws = (float*)d_ws;
  size_t off = 0;
  float*    h1   = ws + off; off += (size_t)Nn * C1;       // x @ W1 (f32)
  float*    out1 = ws + off; off += (size_t)Nn * C1;       // L1 aggregation (f32 atomics)
  float*    h2   = ws + off; off += (size_t)Nn * C2;       // relu(out1+b1) @ W2 (f32)
  float*    out2 = ws + off; off += (size_t)Nn * C2;       // L2 aggregation (f32 atomics)
  float*    as1  = ws + off; off += (size_t)Nn * H1;
  float*    ad1  = ws + off; off += (size_t)Nn * H1;
  unsigned* m1   = (unsigned*)(ws + off); off += (size_t)Nn * H1;
  float*    den1 = ws + off; off += (size_t)Nn * H1;
  float*    as2  = ws + off; off += (size_t)Nn;
  float*    ad2  = ws + off; off += (size_t)Nn;
  unsigned* m2   = (unsigned*)(ws + off); off += (size_t)Nn;
  float*    den2 = ws + off; off += (size_t)Nn;
  float*    ev   = ws + off; off += (size_t)Etot * H1;     // edge scratch (L2 reuses, H=1)
  __bf16*   xbf    = (__bf16*)(ws + off); off += (size_t)Nn * K1 / 2;  // x in bf16
  __bf16*   out1bf = (__bf16*)(ws + off); off += (size_t)Nn * C1 / 2;  // relu(out1+b1) bf16
  __bf16*   W1p    = (__bf16*)(ws + off); off += (size_t)K1 * C1 / 2;  // packed fragments
  __bf16*   W2p    = (__bf16*)(ws + off); off += (size_t)C1 * C2 / 2;

  const int TPB = 256;
  const int WPB = TPB / 32;

  // zero-init accumulators / maxima for both layers
  {
    long long n;
    n = (long long)Nn * C1; fill_u32_kernel<<<cdiv_ll(n, TPB), TPB, 0, stream>>>((unsigned*)out1, 0u, n);
    n = (long long)Nn * C2; fill_u32_kernel<<<cdiv_ll(n, TPB), TPB, 0, stream>>>((unsigned*)out2, 0u, n);
    n = (long long)Nn * H1; fill_u32_kernel<<<cdiv_ll(n, TPB), TPB, 0, stream>>>(m1, 0u, n);
    n = (long long)Nn * H1; fill_u32_kernel<<<cdiv_ll(n, TPB), TPB, 0, stream>>>((unsigned*)den1, 0u, n);
    n = (long long)Nn;      fill_u32_kernel<<<cdiv_ll(n, TPB), TPB, 0, stream>>>(m2, 0u, n);
    n = (long long)Nn;      fill_u32_kernel<<<cdiv_ll(n, TPB), TPB, 0, stream>>>((unsigned*)den2, 0u, n);
  }

  // precompute bf16 A copy and packed-fragment weights
  f32_to_bf16_kernel<<<cdiv_ll((long long)Nn * K1, TPB), TPB, 0, stream>>>(x, xbf, (long long)Nn * K1);
  pack_B_kernel<<<cdiv_ll((K1 / 32) * (C1 / 16) * 32, TPB), TPB, 0, stream>>>(W1, W1p, K1, C1);
  pack_B_kernel<<<cdiv_ll((C1 / 32) * (C2 / 16) * 32, TPB), TPB, 0, stream>>>(W2, W2p, C1, C2);

  const int mtiles = Nn / 16;                       // 3125
  const int gemm_blocks = cdiv_ll(mtiles, WPB);

  // ---- layer 1: h1 = x @ W1 (K=128, N=256, heads=4 concat)
  gemm_bf16_wmma<16><<<gemm_blocks, TPB, 0, stream>>>(xbf, W1p, h1, Nn, C1, K1);
  alpha_reduce_kernel<<<cdiv_ll((long long)Nn * H1, WPB), TPB, 0, stream>>>(h1, a_src1, a_dst1, as1, ad1, Nn, H1);
  edge_pass1_kernel<4><<<cdiv_ll(Etot, TPB), TPB, 0, stream>>>(src, dst, as1, ad1, ev, m1, Eraw, Etot);
  edge_pass2_kernel<4><<<cdiv_ll(Etot, TPB), TPB, 0, stream>>>(dst, ev, m1, den1, Eraw, Etot);
  edge_pass3_kernel<4><<<cdiv_ll(Etot, WPB), TPB, 0, stream>>>(src, dst, ev, den1, h1, out1, Eraw, Etot);
  bias_relu_bf16_kernel<<<cdiv_ll((long long)Nn * C1, TPB), TPB, 0, stream>>>(out1, out1bf, b1, (long long)Nn * C1, C1);

  // ---- layer 2: h2 = relu(out1+b1) @ W2 (K=256, N=64, heads=1)
  gemm_bf16_wmma<4><<<gemm_blocks, TPB, 0, stream>>>(out1bf, W2p, h2, Nn, C2, C1);
  alpha_reduce_kernel<<<cdiv_ll((long long)Nn, WPB), TPB, 0, stream>>>(h2, a_src2, a_dst2, as2, ad2, Nn, 1);
  edge_pass1_kernel<1><<<cdiv_ll(Etot, TPB), TPB, 0, stream>>>(src, dst, as2, ad2, ev, m2, Eraw, Etot);
  edge_pass2_kernel<1><<<cdiv_ll(Etot, TPB), TPB, 0, stream>>>(dst, ev, m2, den2, Eraw, Etot);
  edge_pass3_kernel<1><<<cdiv_ll(Etot, WPB), TPB, 0, stream>>>(src, dst, ev, den2, h2, out2, Eraw, Etot);
  bias_relu_kernel<<<cdiv_ll((long long)Nn * C2, TPB), TPB, 0, stream>>>(out2, b2, (long long)Nn * C2, C2);

  // ---- classifier head
  classifier_kernel<<<cdiv_ll(Nn, TPB), TPB, 0, stream>>>(out2, Wc1, bc1, Wc2, bc2, out, Nn);
}
